// WindowBasedCrossAttention_74217034874926
// MI455X (gfx1250) — matrix operand order repaired and checked by reading, not compiled, not run on hardware
//
#include <hip/hip_runtime.h>

typedef __attribute__((ext_vector_type(16))) _Float16 v16h;
typedef __attribute__((ext_vector_type(8)))  _Float16 v8h;
typedef __attribute__((ext_vector_type(2)))  _Float16 v2h;
typedef __attribute__((ext_vector_type(8)))  float    v8f;
typedef __attribute__((ext_vector_type(4)))  float    v4f;

static constexpr int Dk  = 1024;   // model dim
static constexpr int Sk  = 4096;   // sequence length
static constexpr int WIN = 256;    // band half-width
static constexpr int KT  = 32;     // keys per tile

static __device__ __forceinline__ v16h join16(v8h a, v8h b) {
  return __builtin_shufflevector(a, b, 0,1,2,3,4,5,6,7,8,9,10,11,12,13,14,15);
}

static __device__ __forceinline__ v8f wmma_f16(v16h a, v16h b, v8f c) {
  // D = A(16x32 f16) * B(32x16 f16) + C(16x16 f32)
  return __builtin_amdgcn_wmma_f32_16x16x32_f16(false, a, false, b, (short)0, c,
                                                false, false);
}

// ---------------------------------------------------------------------------
// Kernel 0: one-time f32 -> f16 conversion (used for the weight matrices so
// the GEMM inner loop reads B fragments with zero VALU conversion work).
// ---------------------------------------------------------------------------
__global__ __launch_bounds__(256) void cvt_f32_f16(const float* __restrict__ src,
                                                   _Float16* __restrict__ dst) {
  const size_t i = ((size_t)blockIdx.x * 256 + threadIdx.x) * 8;
  v4f a = *(const v4f*)(src + i);
  v4f b = *(const v4f*)(src + i + 4);
  v8h o;
#pragma unroll
  for (int e = 0; e < 4; ++e) {
    o[e]     = (_Float16)a[e];
    o[4 + e] = (_Float16)b[e];
  }
  *(v8h*)(dst + i) = o;
}

// ---------------------------------------------------------------------------
// Kernel 1: Out[row][e] = sum_d In[row][d] * Wh[e][d]   (f32 A, f16 B, f16 out)
// Block = 256 threads = 8 waves. Block tile: 16 rows x 512 cols.
// Wave tile: 16 x 64 (four 16x16 accumulators). K-loop in chunks of 32.
// ---------------------------------------------------------------------------
__global__ __launch_bounds__(256) void qkv_proj_f16(const float* __restrict__ In,
                                                    const _Float16* __restrict__ Wh,
                                                    _Float16* __restrict__ Out) {
  const int tid  = threadIdx.x;
  const int wave = tid >> 5;
  const int lane = tid & 31;
  const int n    = lane & 15;
  const int hi   = lane >> 4;
  const int b8   = hi << 3;            // A-fragment K base
  const int o16  = hi << 4;            // unused; B fragments also use b8

  const int mbase = blockIdx.x * 16;
  const int nb    = blockIdx.y * 512 + wave * 64;

  const float* __restrict__ ap = In + (size_t)(mbase + n) * Dk;
  const _Float16* bp[4];
#pragma unroll
  for (int f = 0; f < 4; ++f)
    bp[f] = Wh + (size_t)(nb + f * 16 + n) * Dk;

  v8f acc[4];
#pragma unroll
  for (int f = 0; f < 4; ++f) acc[f] = v8f{};

#pragma unroll 2
  for (int kc = 0; kc < Dk; kc += 32) {
    // ---- A fragment (f32 -> f16): K runs [b8,b8+8) and [b8+16,b8+24)
    v4f a0 = *(const v4f*)(ap + kc + b8);
    v4f a1 = *(const v4f*)(ap + kc + b8 + 4);
    v4f a2 = *(const v4f*)(ap + kc + b8 + 16);
    v4f a3 = *(const v4f*)(ap + kc + b8 + 20);
    v16h A;
#pragma unroll
    for (int i = 0; i < 4; ++i) {
      A[i]      = (_Float16)a0[i];
      A[4 + i]  = (_Float16)a1[i];
      A[8 + i]  = (_Float16)a2[i];
      A[12 + i] = (_Float16)a3[i];
    }
    // ---- B fragments: raw f16 b128 loads, no conversion
#pragma unroll
    for (int f = 0; f < 4; ++f) {
      v16h B = join16(*(const v8h*)(bp[f] + kc + b8),
                      *(const v8h*)(bp[f] + kc + b8 + 16));
      acc[f] = wmma_f16(A, B, acc[f]);
    }
  }

  // ---- Store C fragments as f16 (lanes 0-15: rows mbase+r; 16-31: +8)
#pragma unroll
  for (int f = 0; f < 4; ++f) {
#pragma unroll
    for (int r = 0; r < 8; ++r) {
      const size_t row = (size_t)(mbase + r + hi * 8);
      Out[row * Dk + nb + f * 16 + n] = (_Float16)acc[f][r];
    }
  }
}

// ---------------------------------------------------------------------------
// Kernel 2: banded flash attention over precomputed f16 Q,K,V.
// Block = 256 threads = 8 waves, handles 16 queries of one batch.
// Wave w owns output columns [w*128, w*128+128): 8 f32 accumulators.
// ---------------------------------------------------------------------------
__global__ __launch_bounds__(256) void band_attn(const _Float16* __restrict__ Qg,
                                                 const _Float16* __restrict__ Kg,
                                                 const _Float16* __restrict__ Vg,
                                                 float* __restrict__ O) {
  __shared__ __align__(16) _Float16 Vt[Dk * KT];     // V tile transposed [d][key]
  __shared__ __align__(16) float    Ssc[16 * KT];    // score tile  [m][j]
  __shared__ __align__(16) _Float16 Pt[16 * KT];     // prob  tile  [m][j]
  __shared__ __align__(16) float    rowmax[16];
  __shared__ __align__(16) float    rowsum[16];
  __shared__ __align__(16) float    alphaS[16];

  const int tid  = threadIdx.x;
  const int wave = tid >> 5;
  const int lane = tid & 31;
  const int n    = lane & 15;
  const int hi   = lane >> 4;
  const int b8   = hi << 3;
  const int o16  = hi << 4;

  const int qbase = blockIdx.x * 16;
  const int b     = blockIdx.y;
  const int dbase = wave * 128;
  const float scale = 0.03125f;               // 1/sqrt(1024)

  if (tid < 16) { rowmax[tid] = -1e30f; rowsum[tid] = 0.0f; }
  __syncthreads();

  v8f acc[8];
#pragma unroll
  for (int f = 0; f < 8; ++f) acc[f] = v8f{};

  int jstart = qbase - WIN; if (jstart < 0) jstart = 0;
  jstart &= ~(KT - 1);
  int jend = qbase + 16 + WIN; if (jend > Sk) jend = Sk;

  for (int kb = jstart; kb < jend; kb += KT) {
    // ---- (A) zero score tile + stage V tile transposed into LDS ----------
    for (int i = tid; i < 16 * KT; i += 256) Ssc[i] = 0.0f;
    {
      // Thread handles a pair of adjacent keys x 64 d values: packed b32
      // stores into the transposed tile.
      const int key0 = (tid >> 4) * 2;                 // 0,2,..,30
      const int dseg = (tid & 15) * 64;
      int kr0 = kb + key0;     if (kr0 > Sk - 1) kr0 = Sk - 1;
      int kr1 = kb + key0 + 1; if (kr1 > Sk - 1) kr1 = Sk - 1;
      const _Float16* vp0 = Vg + ((size_t)(b * Sk + kr0)) * Dk + dseg;
      const _Float16* vp1 = Vg + ((size_t)(b * Sk + kr1)) * Dk + dseg;
#pragma unroll
      for (int u = 0; u < 8; ++u) {
        v8h va = *(const v8h*)(vp0 + u * 8);
        v8h vb = *(const v8h*)(vp1 + u * 8);
        const int d0 = dseg + u * 8;
#pragma unroll
        for (int e = 0; e < 8; ++e) {
          v2h pr = {va[e], vb[e]};
          *(v2h*)&Vt[(d0 + e) * KT + key0] = pr;
        }
      }
    }
    __syncthreads();

    // ---- (B) partial scores over this wave's D slice, LDS-reduce ---------
    {
      v8f s0 = {}, s1 = {};
      const int qrow = qbase + n;
      const _Float16* qp = Qg + ((size_t)(b * Sk + qrow)) * Dk;
      int k0 = kb + n;      if (k0 > Sk - 1) k0 = Sk - 1;
      int k1 = kb + 16 + n; if (k1 > Sk - 1) k1 = Sk - 1;
      const _Float16* kp0 = Kg + ((size_t)(b * Sk + k0)) * Dk;
      const _Float16* kp1 = Kg + ((size_t)(b * Sk + k1)) * Dk;
      // Prefetch next tile's K rows (global_prefetch_b8) to hide the
      // strided row accesses across the kb loop.
      __builtin_prefetch(kp0 + (size_t)KT * Dk, 0, 0);
      __builtin_prefetch(kp1 + (size_t)KT * Dk, 0, 0);
#pragma unroll
      for (int c = 0; c < 4; ++c) {
        const int kc = dbase + c * 32;
        v16h A  = join16(*(const v8h*)(qp  + kc + b8),
                         *(const v8h*)(qp  + kc + b8 + 16));
        v16h B0 = join16(*(const v8h*)(kp0 + kc + b8),
                         *(const v8h*)(kp0 + kc + b8 + 16));
        v16h B1 = join16(*(const v8h*)(kp1 + kc + b8),
                         *(const v8h*)(kp1 + kc + b8 + 16));
        s0 = wmma_f16(A, B0, s0);
        s1 = wmma_f16(A, B1, s1);
      }
#pragma unroll
      for (int r = 0; r < 8; ++r) {
        const int m = r + hi * 8;
        atomicAdd(&Ssc[m * KT + n],      s0[r] * scale);
        atomicAdd(&Ssc[m * KT + 16 + n], s1[r] * scale);
      }
    }
    __syncthreads();

    // ---- (C) online softmax update (16 threads, one row each) ------------
    if (tid < 16) {
      const int m = tid;
      const int q = qbase + m;
      float sv[KT];
      float tmax = -1e30f;
#pragma unroll
      for (int j = 0; j < KT; ++j) {
        const int key = kb + j;
        int diff = key - q; if (diff < 0) diff = -diff;
        float s = Ssc[m * KT + j];
        if (key >= Sk || diff > WIN) s = -1e10f;
        sv[j] = s;
        tmax = fmaxf(tmax, s);
      }
      const float mo   = rowmax[m];
      const float newm = fmaxf(mo, tmax);
      const float al   = __expf(mo - newm);
      float psum = 0.0f;
#pragma unroll
      for (int j = 0; j < KT; ++j) {
        const float p = __expf(sv[j] - newm);
        psum += p;
        Pt[m * KT + j] = (_Float16)p;
      }
      rowmax[m] = newm;
      rowsum[m] = rowsum[m] * al + psum;
      alphaS[m] = al;
    }
    __syncthreads();

    // ---- (D) rescale accumulators, then P x V over this wave's slice -----
    {
      v4f al0 = *(const v4f*)&alphaS[b8];
      v4f al1 = *(const v4f*)&alphaS[b8 + 4];
#pragma unroll
      for (int f = 0; f < 8; ++f)
#pragma unroll
        for (int r = 0; r < 8; ++r)
          acc[f][r] *= (r < 4 ? al0[r] : al1[r - 4]);

      const _Float16* pp = Pt + n * KT;
      v16h AP = join16(*(const v8h*)(pp + b8),
                       *(const v8h*)(pp + b8 + 16));
#pragma unroll
      for (int f = 0; f < 8; ++f) {
        const int d = dbase + f * 16 + n;
        const _Float16* vb = Vt + d * KT + b8;
        v16h BV = join16(*(const v8h*)(vb), *(const v8h*)(vb + 16));
        acc[f] = wmma_f16(AP, BV, acc[f]);
      }
    }
    __syncthreads();   // protect Vt/Pt/Ssc for the next tile
  }

  // ---- finalize: divide by row sums and store f32 -------------------------
  v4f l0 = *(const v4f*)&rowsum[b8];
  v4f l1 = *(const v4f*)&rowsum[b8 + 4];
#pragma unroll
  for (int f = 0; f < 8; ++f) {
#pragma unroll
    for (int r = 0; r < 8; ++r) {
      const float ls  = (r < 4 ? l0[r] : l1[r - 4]);
      const size_t row = (size_t)(b * Sk + qbase + r + hi * 8);
      O[row * Dk + dbase + f * 16 + n] = acc[f][r] / ls;
    }
  }
}

// ---------------------------------------------------------------------------
extern "C" void kernel_launch(void* const* d_in, const int* in_sizes, int n_in,
                              void* d_out, int out_size, void* d_ws, size_t ws_size,
                              hipStream_t stream) {
  const float* X  = (const float*)d_in[0];
  const float* Y  = (const float*)d_in[1];
  const float* Wq = (const float*)d_in[2];
  const float* Wk = (const float*)d_in[3];
  const float* Wv = (const float*)d_in[4];
  float* Out = (float*)d_out;

  const int B    = in_sizes[0] / (Sk * Dk);   // = 4
  const int rows = B * Sk;                    // = 16384

  _Float16* Qg  = (_Float16*)d_ws;
  _Float16* Kg  = Qg + (size_t)rows * Dk;
  _Float16* Vg  = Kg + (size_t)rows * Dk;
  _Float16* Wqh = Vg + (size_t)rows * Dk;
  _Float16* Wkh = Wqh + (size_t)Dk * Dk;
  _Float16* Wvh = Wkh + (size_t)Dk * Dk;

  // One-time weight conversion: D*D / (256 threads * 8 elems) blocks.
  const int cvtBlocks = (Dk * Dk) / (256 * 8);   // 512
  cvt_f32_f16<<<cvtBlocks, 256, 0, stream>>>(Wq, Wqh);
  cvt_f32_f16<<<cvtBlocks, 256, 0, stream>>>(Wk, Wkh);
  cvt_f32_f16<<<cvtBlocks, 256, 0, stream>>>(Wv, Wvh);

  dim3 gProj(rows / 16, Dk / 512);
  qkv_proj_f16<<<gProj, 256, 0, stream>>>(Y, Wqh, Qg);   // Q = Y @ Wq^T
  qkv_proj_f16<<<gProj, 256, 0, stream>>>(X, Wkh, Kg);   // K = X @ Wk^T
  qkv_proj_f16<<<gProj, 256, 0, stream>>>(X, Wvh, Vg);   // V = X @ Wv^T

  dim3 gAttn(Sk / 16, B);
  band_attn<<<gAttn, 256, 0, stream>>>(Qg, Kg, Vg, Out);
}
